// FullAttention_62603443306552
// MI455X (gfx1250) — matrix operand order repaired and checked by reading, not compiled, not run on hardware
//
#include <hip/hip_runtime.h>

typedef _Float16 v4h  __attribute__((ext_vector_type(4)));
typedef _Float16 v8h  __attribute__((ext_vector_type(8)));
typedef _Float16 v16h __attribute__((ext_vector_type(16)));
typedef float    v8f  __attribute__((ext_vector_type(8)));
typedef unsigned int v8u __attribute__((ext_vector_type(8)));

#define B_  4
#define L_  2048
#define H_  16
#define E_  64
#define S_  2048
#define D_  64
#define BQ  128         // query rows per block (4 waves x 2 tiles x 16)
#define KB  64          // keys per staged LDS tile (softmax granularity)
#define LDK 72          // K LDS row stride in halves
#define LDV 36          // Vt LDS row stride in dwords (2 keys packed per dword)

// (1/sqrt(E)) * log2(e): softmax computed in exp2 domain, scale folded into Q
#define QSCALE 0.18033688011112042f

static __device__ __forceinline__ v16h cat8(v8h lo, v8h hi) {
  return __builtin_shufflevector(lo, hi, 0,1,2,3,4,5,6,7,8,9,10,11,12,13,14,15);
}

// lane <-> lane^16 exchange on the VALU pipe (no LDS traffic)
static __device__ __forceinline__ unsigned xhalf_u(unsigned v) {
#if __has_builtin(__builtin_amdgcn_permlanex16)
  return (unsigned)__builtin_amdgcn_permlanex16((int)v, (int)v,
                                                0x76543210, (int)0xFEDCBA98,
                                                false, false);
#else
  return (unsigned)__shfl_xor((int)v, 16, 32);
#endif
}
static __device__ __forceinline__ float xhalf_f(float v) {
  return __builtin_bit_cast(float, xhalf_u(__builtin_bit_cast(unsigned, v)));
}

static __device__ __forceinline__ float fexp2(float x) {
#if __has_builtin(__builtin_amdgcn_exp2f)
  return __builtin_amdgcn_exp2f(x);   // raw v_exp_f32
#else
  return __builtin_exp2f(x);
#endif
}

static __device__ __forceinline__ unsigned pk16(float lo, float hi) {
  return __builtin_bit_cast(unsigned, __builtin_amdgcn_cvt_pkrtz(lo, hi));
}

#define WMMA_RAB(A, Bm, C, RA, RB) \
  __builtin_amdgcn_wmma_f32_16x16x32_f16(false, (A), false, (Bm), (short)0, (C), (RA), (RB))

__global__ __launch_bounds__(128, 2)
void fa_causal_wmma(const float* __restrict__ Qg,
                    const float* __restrict__ Kg,
                    const float* __restrict__ Vg,
                    float* __restrict__ Og)
{
  __shared__ _Float16 Klds[KB * LDK];   // [key][e] f16
  __shared__ unsigned VtU [D_ * LDV];   // [d][key/2] packed f16x2 (transposed)

  const int qb   = blockIdx.x;          // 0..15 (128 query rows each)
  const int h    = blockIdx.y;
  const int b    = blockIdx.z;
  const int tid  = threadIdx.x;
  const int lane = tid & 31;
  const int wave = tid >> 5;
  const int lm   = lane & 15;
  const int hl   = lane >> 4;

  const int m0a = qb * BQ + wave * 16;  // query tile A
  const int m0b = m0a + 64;             // query tile B

  // ---- Q for both tiles, B-operand layout for S^T = K*Q^T ----
  v16h qB[2][2];
  #pragma unroll
  for (int u = 0; u < 2; ++u) {
    const int qr = (u ? m0b : m0a) + lm;
    const float4* q4 = (const float4*)(Qg + (((size_t)b * L_ + qr) * H_ + h) * E_);
    #pragma unroll
    for (int j = 0; j < 4; ++j) {
      float4 a = q4[hl * 4 + j];
      float4 c = q4[8 + hl * 4 + j];
      qB[u][0][4*j+0] = (_Float16)(a.x * QSCALE);
      qB[u][0][4*j+1] = (_Float16)(a.y * QSCALE);
      qB[u][0][4*j+2] = (_Float16)(a.z * QSCALE);
      qB[u][0][4*j+3] = (_Float16)(a.w * QSCALE);
      qB[u][1][4*j+0] = (_Float16)(c.x * QSCALE);
      qB[u][1][4*j+1] = (_Float16)(c.y * QSCALE);
      qB[u][1][4*j+2] = (_Float16)(c.z * QSCALE);
      qB[u][1][4*j+3] = (_Float16)(c.w * QSCALE);
    }
  }

  // ---- staging pointers ----
  const int srow = tid >> 4;     // 0..7
  const int c4   = tid & 15;     // float4 column
  const float* kSrc = Kg + (((size_t)b * S_ + srow) * H_ + h) * E_ + c4 * 4;
  const float* vSrc = Vg + (((size_t)b * S_) * H_ + h) * D_ + c4 * 4;
  const size_t rowK8 = (size_t)8 * H_ * E_;     // 8 key rows (K)
  const size_t rowV  = (size_t)H_ * D_;         // 1 key row (V)
  const size_t blkS  = (size_t)KB * H_ * E_;    // 64 key rows (E_==D_)
  _Float16* kDst = &Klds[srow * LDK + c4 * 4];

  const float NEG_INF = -__builtin_inff();
  v8f acc0[4] = {}, acc1[4] = {};
  float m_a = NEG_INF, l_a = 0.0f;
  float m_b = NEG_INF, l_b = 0.0f;

  // A-operand from K LDS (16x32 f16): elem i -> k=hl*16+i
  auto kA = [&](int t, int e) -> v16h {
    const _Float16* p = &Klds[(t * 16 + lm) * LDK + e + hl * 16];
    return cat8(*(const v8h*)p, *(const v8h*)(p + 8));
  };
  // A-operand from packed V^T LDS: 4+4 consecutive key-dwords
  auto vA = [&](int dt, int ch) -> v16h {
    const unsigned* p = &VtU[(dt * 16 + lm) * LDV + ch * 16 + hl * 8];
    return cat8(*(const v8h*)p, *(const v8h*)(p + 4));
  };
  auto mask_tail = [&](v8f (&st)[4]) {
    const int qoff = wave * 16 + lm;
    #pragma unroll
    for (int t = 0; t < 4; ++t)
      #pragma unroll
      for (int r = 0; r < 8; ++r)
        if (t * 16 + hl * 8 + r > qoff) st[t][r] = NEG_INF;
  };
  auto softmax_update = [&](v8f (&st)[4], float& m_i, float& l_i) -> float {
    float tmax = NEG_INF;
    #pragma unroll
    for (int t = 0; t < 4; ++t)
      #pragma unroll
      for (int r = 0; r < 8; ++r) tmax = fmaxf(tmax, st[t][r]);
    tmax = fmaxf(tmax, xhalf_f(tmax));
    const float m_new = fmaxf(m_i, tmax);
    const float alpha = fexp2(m_i - m_new);
    float rsum = 0.0f;
    #pragma unroll
    for (int t = 0; t < 4; ++t)
      #pragma unroll
      for (int r = 0; r < 8; ++r) {
        const float p = fexp2(st[t][r] - m_new);
        st[t][r] = p;
        rsum += p;
      }
    rsum += xhalf_f(rsum);
    l_i = l_i * alpha + rsum;
    m_i = m_new;
    return alpha;
  };
  auto buildPB = [&](const v8f (&st)[4], int ch) -> v16h {
    v8u d;
    #pragma unroll
    for (int j = 0; j < 4; ++j) {
      unsigned c0 = pk16(st[2*ch][2*j],   st[2*ch][2*j+1]);
      unsigned c1 = pk16(st[2*ch+1][2*j], st[2*ch+1][2*j+1]);
      unsigned w = hl ? c0 : c1;
      unsigned x = xhalf_u(w);
      d[j]     = hl ? x  : c0;
      d[4 + j] = hl ? c1 : x;
    }
    return __builtin_bit_cast(v16h, d);
  };

  const int kbmax = 2 * qb + 1;
  for (int kb = 0; kb <= kbmax; ++kb) {
    __syncthreads();
    // ---- stage K (f16 rows) ----
    const float* kS = kSrc + (size_t)kb * blkS;
    #pragma unroll
    for (int it = 0; it < 8; ++it) {
      float4 kk = *(const float4*)(kS + it * rowK8);
      v4h kh = { (_Float16)kk.x, (_Float16)kk.y, (_Float16)kk.z, (_Float16)kk.w };
      *(v4h*)(kDst + it * 8 * LDK) = kh;
    }
    // ---- stage V transposed, key-pairs packed into dwords ----
    const float* vS = vSrc + (size_t)kb * blkS;
    #pragma unroll
    for (int it = 0; it < 4; ++it) {
      const int p = srow + it * 8;                       // key pair index 0..31
      const float* rp = vS + (size_t)(2 * p) * rowV;
      float4 lo = *(const float4*)rp;
      float4 hi = *(const float4*)(rp + rowV);
      VtU[(c4*4+0) * LDV + p] = pk16(lo.x, hi.x);
      VtU[(c4*4+1) * LDV + p] = pk16(lo.y, hi.y);
      VtU[(c4*4+2) * LDV + p] = pk16(lo.z, hi.z);
      VtU[(c4*4+3) * LDV + p] = pk16(lo.w, hi.w);
    }
    __syncthreads();

    const bool doA = (kb <= 2 * qb);     // tile A still needs keys
    v8f st0[4], st1[4];
    v8f z = {};

    if (doA) {  // interleaved: each K A-operand feeds both query tiles (reuse_a)
      { v16h a = kA(0,0);  st0[0] = WMMA_RAB(a, qB[0][0], z, false, false);
                           st1[0] = WMMA_RAB(a, qB[1][0], z, true,  false); }
      { v16h a = kA(1,0);  st0[1] = WMMA_RAB(a, qB[0][0], z, false, false);
                           st1[1] = WMMA_RAB(a, qB[1][0], z, true,  false); }
      { v16h a = kA(2,0);  st0[2] = WMMA_RAB(a, qB[0][0], z, false, false);
                           st1[2] = WMMA_RAB(a, qB[1][0], z, true,  false); }
      { v16h a = kA(3,0);  st0[3] = WMMA_RAB(a, qB[0][0], z, false, false);
                           st1[3] = WMMA_RAB(a, qB[1][0], z, true,  false); }
      { v16h a = kA(0,32); st0[0] = WMMA_RAB(a, qB[0][1], st0[0], false, false);
                           st1[0] = WMMA_RAB(a, qB[1][1], st1[0], true,  false); }
      { v16h a = kA(1,32); st0[1] = WMMA_RAB(a, qB[0][1], st0[1], false, false);
                           st1[1] = WMMA_RAB(a, qB[1][1], st1[1], true,  false); }
      { v16h a = kA(2,32); st0[2] = WMMA_RAB(a, qB[0][1], st0[2], false, false);
                           st1[2] = WMMA_RAB(a, qB[1][1], st1[2], true,  false); }
      { v16h a = kA(3,32); st0[3] = WMMA_RAB(a, qB[0][1], st0[3], false, false);
                           st1[3] = WMMA_RAB(a, qB[1][1], st1[3], true,  false); }
    } else {    // only tile B: share B operand across tiles (reuse_b)
      st1[0] = WMMA_RAB(kA(0,0),  qB[1][0], z, false, false);
      st1[1] = WMMA_RAB(kA(1,0),  qB[1][0], z, false, true);
      st1[2] = WMMA_RAB(kA(2,0),  qB[1][0], z, false, true);
      st1[3] = WMMA_RAB(kA(3,0),  qB[1][0], z, false, true);
      st1[0] = WMMA_RAB(kA(0,32), qB[1][1], st1[0], false, false);
      st1[1] = WMMA_RAB(kA(1,32), qB[1][1], st1[1], false, true);
      st1[2] = WMMA_RAB(kA(2,32), qB[1][1], st1[2], false, true);
      st1[3] = WMMA_RAB(kA(3,32), qB[1][1], st1[3], false, true);
    }

    if (kb == 2 * qb) mask_tail(st0);    // tile A diagonal
    if (kb == kbmax)  mask_tail(st1);    // tile B diagonal

    const float alB = softmax_update(st1, m_b, l_b);
    #pragma unroll
    for (int dt = 0; dt < 4; ++dt)
      #pragma unroll
      for (int r = 0; r < 8; ++r) acc1[dt][r] *= alB;

    if (doA) {
      const float alA = softmax_update(st0, m_a, l_a);
      #pragma unroll
      for (int dt = 0; dt < 4; ++dt)
        #pragma unroll
        for (int r = 0; r < 8; ++r) acc0[dt][r] *= alA;
    }

    #pragma unroll
    for (int ch = 0; ch < 2; ++ch) {
      if (doA) {
        const v16h pBA = buildPB(st0, ch);
        const v16h pBB = buildPB(st1, ch);
        { v16h a = vA(0,ch); acc0[0] = WMMA_RAB(a, pBA, acc0[0], false, false);
                             acc1[0] = WMMA_RAB(a, pBB, acc1[0], true,  false); }
        { v16h a = vA(1,ch); acc0[1] = WMMA_RAB(a, pBA, acc0[1], false, false);
                             acc1[1] = WMMA_RAB(a, pBB, acc1[1], true,  false); }
        { v16h a = vA(2,ch); acc0[2] = WMMA_RAB(a, pBA, acc0[2], false, false);
                             acc1[2] = WMMA_RAB(a, pBB, acc1[2], true,  false); }
        { v16h a = vA(3,ch); acc0[3] = WMMA_RAB(a, pBA, acc0[3], false, false);
                             acc1[3] = WMMA_RAB(a, pBB, acc1[3], true,  false); }
      } else {
        const v16h pBB = buildPB(st1, ch);
        acc1[0] = WMMA_RAB(vA(0,ch), pBB, acc1[0], false, false);
        acc1[1] = WMMA_RAB(vA(1,ch), pBB, acc1[1], false, true);
        acc1[2] = WMMA_RAB(vA(2,ch), pBB, acc1[2], false, true);
        acc1[3] = WMMA_RAB(vA(3,ch), pBB, acc1[3], false, true);
      }
    }
  }

  // ---- normalize, write O[b][row][h][d], d = dt*16 + 8*hl + r ----
  auto writeO = [&](const v8f (&a)[4], float l, int mrow) {
    const float inv = 1.0f / l;
    float* optr = Og + (((size_t)b * L_ + mrow + lm) * H_ + h) * D_;
    #pragma unroll
    for (int dt = 0; dt < 4; ++dt) {
      float4 o0 = { a[dt][0]*inv, a[dt][1]*inv, a[dt][2]*inv, a[dt][3]*inv };
      float4 o1 = { a[dt][4]*inv, a[dt][5]*inv, a[dt][6]*inv, a[dt][7]*inv };
      *(float4*)(optr + dt * 16 + hl * 8)     = o0;
      *(float4*)(optr + dt * 16 + hl * 8 + 4) = o1;
    }
  };
  writeO(acc0, l_a, m0a);
  writeO(acc1, l_b, m0b);
}

extern "C" void kernel_launch(void* const* d_in, const int* in_sizes, int n_in,
                              void* d_out, int out_size, void* d_ws, size_t ws_size,
                              hipStream_t stream) {
  (void)in_sizes; (void)n_in; (void)d_ws; (void)ws_size; (void)out_size;
  const float* Q = (const float*)d_in[0];
  const float* K = (const float*)d_in[1];
  const float* V = (const float*)d_in[2];
  float* O = (float*)d_out;
  dim3 grid(L_ / BQ, H_, B_);   // (16, 16, 4)
  fa_causal_wmma<<<grid, 128, 0, stream>>>(Q, K, V, O);
}
